// CausalAttention_43903155700334
// MI455X (gfx1250) — matrix operand order repaired and checked by reading, not compiled, not run on hardware
//
#include <hip/hip_runtime.h>
#include <math.h>

// ---------------------------------------------------------------------------
// CDNA5 (gfx1250) fused causal attention, B=4 T=2048 D=1024, fp32 in/out.
// fp32 -> bf16 once, all GEMMs via v_wmma_f32_16x16x32_bf16 (fp32 accum),
// flash-style online softmax.
// ---------------------------------------------------------------------------

typedef __attribute__((ext_vector_type(16))) __bf16 v16bf;
typedef __attribute__((ext_vector_type(8)))  float  v8f;
typedef __attribute__((ext_vector_type(4)))  unsigned int v4u;
typedef __attribute__((ext_vector_type(2)))  unsigned int v2u;

union FragAB {            // 16 bf16 elements = 32 bytes = 8 VGPRs
    v4u   u4[2];
    v16bf v;
};

// Native bf16 conversion: single v_cvt (RNE) instead of a 4-op integer RNE.
__device__ __forceinline__ unsigned short f2bf(float f) {
    union { __bf16 h; unsigned short u; } c;
    c.h = (__bf16)f;
    return c.u;
}
__device__ __forceinline__ unsigned pk2bf(float a, float b) {
    union { __bf16 h[2]; unsigned u; } c;
    c.h[0] = (__bf16)a;
    c.h[1] = (__bf16)b;
    return c.u;
}

#define BT   8192          // B*T
#define DD   1024
#define TT   2048

// ---------------------------------------------------------------------------
// Kernel 1: out = x @ W for W in {Wq,Wk,Wv}; bf16 outputs. z==2 stores V
// transposed as Vt[b][d][t] so the attention kernel's B fragments are
// contiguous.
// Block tile: 128(M) x 128(N), K-step 32. 8 waves -> 4x2 wave grid, each wave
// computes a 32x64 macro tile (2x4 wmma accumulators -> 8 WMMAs / k-step).
// ---------------------------------------------------------------------------
__global__ __launch_bounds__(256)
void qkv_gemm_kernel(const float* __restrict__ x,
                     const float* __restrict__ Wq,
                     const float* __restrict__ Wk,
                     const float* __restrict__ Wv,
                     unsigned short* __restrict__ Qb,
                     unsigned short* __restrict__ Kb,
                     unsigned short* __restrict__ Vt) {
    const int z = blockIdx.z;
    const float* __restrict__ W = (z == 0) ? Wq : (z == 1) ? Wk : Wv;
    unsigned short* __restrict__ outp = (z == 0) ? Qb : (z == 1) ? Kb : Vt;

    const int m0  = blockIdx.x * 128;
    const int n0  = blockIdx.y * 128;
    const int tid = threadIdx.x;
    const int ln  = tid & 31;
    const int w   = tid >> 5;
    const int lh  = ln & 15;          // lane within half
    const int hi  = ln >> 4;          // which half of the wave
    const int mw  = w & 3;            // wave M index (0..3)
    const int nw  = w >> 2;           // wave N index (0..1)

    // stride 40 halves = 80 bytes -> every row 16B aligned for ds_load_b128
    __shared__ unsigned short lA[128 * 40];   // A tile  [row][k]
    __shared__ unsigned short lB[128 * 40];   // B tile transposed [n][k]

    v8f acc[2][4];
    for (int a = 0; a < 2; ++a)
        for (int b = 0; b < 4; ++b)
            acc[a][b] = (v8f)0.0f;

    const int koffA = hi ? 8 : 0;
    const int koffB = hi ? 8 : 0;

    for (int k0 = 0; k0 < DD; k0 += 32) {
        // stage A: 128 rows x 32 k, float4 loads, packed bf16 b64 stores
        for (int c4 = tid; c4 < 128 * 32 / 4; c4 += 256) {
            int r = c4 >> 3, c = (c4 & 7) * 4;
            float4 f = *(const float4*)(x + (size_t)(m0 + r) * DD + k0 + c);
            v2u p; p.x = pk2bf(f.x, f.y); p.y = pk2bf(f.z, f.w);
            *(v2u*)(&lA[r * 40 + c]) = p;
        }
        // stage Bt: W[k][n] -> lB[n][k]; float4 loads along n, b16 scatter
        for (int c4 = tid; c4 < 128 * 32 / 4; c4 += 256) {
            int kk = c4 >> 5, nn = (c4 & 31) * 4;
            float4 f = *(const float4*)(W + (size_t)(k0 + kk) * DD + n0 + nn);
            lB[(nn + 0) * 40 + kk] = f2bf(f.x);
            lB[(nn + 1) * 40 + kk] = f2bf(f.y);
            lB[(nn + 2) * 40 + kk] = f2bf(f.z);
            lB[(nn + 3) * 40 + kk] = f2bf(f.w);
        }
        __syncthreads();

        FragAB a[2], bfr[4];
        for (int mt = 0; mt < 2; ++mt) {
            const unsigned short* p = &lA[(mw * 32 + mt * 16 + lh) * 40];
            a[mt].u4[0] = *(const v4u*)(p + koffA);
            a[mt].u4[1] = *(const v4u*)(p + 16 + koffA);
        }
        for (int nt = 0; nt < 4; ++nt) {
            const unsigned short* p = &lB[(nw * 64 + nt * 16 + lh) * 40 + koffB];
            bfr[nt].u4[0] = *(const v4u*)(p);
            bfr[nt].u4[1] = *(const v4u*)(p + 16);
        }
        for (int mt = 0; mt < 2; ++mt)
            for (int nt = 0; nt < 4; ++nt)
                acc[mt][nt] = __builtin_amdgcn_wmma_f32_16x16x32_bf16(
                    false, a[mt].v, false, bfr[nt].v,
                    (short)0, acc[mt][nt], false, false);
        __syncthreads();
    }

    // store bf16 results (C layout: VGPR i -> row i + 8*hi, col = lane%16).
    // z is workgroup-uniform: branch hoisted around the whole store loop.
    if (z == 2) {
        for (int mt = 0; mt < 2; ++mt)
            for (int nt = 0; nt < 4; ++nt)
                for (int i = 0; i < 8; ++i) {
                    int m = m0 + mw * 32 + mt * 16 + i + 8 * hi;
                    int n = n0 + nw * 64 + nt * 16 + lh;
                    int bb = m >> 11, t = m & (TT - 1);
                    outp[((size_t)bb * DD + n) * TT + t] = f2bf(acc[mt][nt][i]);
                }
    } else {
        for (int mt = 0; mt < 2; ++mt)
            for (int nt = 0; nt < 4; ++nt)
                for (int i = 0; i < 8; ++i) {
                    int m = m0 + mw * 32 + mt * 16 + i + 8 * hi;
                    int n = n0 + nw * 64 + nt * 16 + lh;
                    outp[(size_t)m * DD + n] = f2bf(acc[mt][nt][i]);
                }
    }
}

// ---------------------------------------------------------------------------
// Kernel 2: flash-style causal attention.
// One WG of 8 waves per 16 query rows. Wave w owns output d-slice
// [w*128, w*128+128). S = Q.K^T is computed as per-wave partials over each
// wave's d-slice and reduced across waves with ds_add_f32 in LDS. Online
// softmax with intra-half shfl reductions; P (bf16, LDS) feeds O += P.V via
// WMMA against the pre-transposed V.
// ---------------------------------------------------------------------------
__global__ __launch_bounds__(256)
void causal_attn_kernel(const unsigned short* __restrict__ Qb,
                        const unsigned short* __restrict__ Kb,
                        const unsigned short* __restrict__ Vt,
                        float* __restrict__ out) {
    const int blk = blockIdx.x;        // 0..511
    const int b   = blk >> 7;          // 128 row-blocks per batch
    const int q0  = (blk & 127) << 4;  // local query start (16 rows)
    const int tid = threadIdx.x;
    const int ln  = tid & 31;
    const int w   = tid >> 5;
    const int lh  = ln & 15;
    const int hi  = ln >> 4;

    __shared__ float          Sred[16][68];    // cross-wave score reduction
    __shared__ unsigned short Pb[16][80];      // bf16 P tile, rows 16B aligned

    const unsigned short* __restrict__ Qrow = Qb + (size_t)(b * TT + q0) * DD;
    const unsigned short* __restrict__ Kbas = Kb + (size_t)b * TT * DD;
    const unsigned short* __restrict__ Vbas = Vt + (size_t)b * DD * TT;

    v8f o[8];
    for (int i = 0; i < 8; ++i) o[i] = (v8f)0.0f;
    float mrow[8], lrow[8];
    for (int i = 0; i < 8; ++i) { mrow[i] = -__builtin_inff(); lrow[i] = 0.0f; }

    const int d0w   = w * 128;
    const int numkb = (q0 + 15) / 64 + 1;   // causal: key blocks of 64
    const float scale = 0.03125f;           // 1/sqrt(1024)
    const int koffA = hi ? 8 : 0;
    const int koffB = hi ? 8 : 0;

    for (int kb = 0; kb < numkb; ++kb) {
        const int key0 = kb * 64;

        // ---- zero the reduction buffer ---------------------------------
        for (int e = tid; e < 16 * 68; e += 256) (&Sred[0][0])[e] = 0.0f;
        __syncthreads();

        // ---- partial S over this wave's d-slice ------------------------
        v8f s[4];
        for (int nt = 0; nt < 4; ++nt) s[nt] = (v8f)0.0f;
        for (int ds = 0; ds < 4; ++ds) {
            const int dd = d0w + ds * 32;
            FragAB qa;
            const unsigned short* qp = Qrow + (size_t)lh * DD + dd;
            qa.u4[0] = *(const v4u*)(qp + koffA);
            qa.u4[1] = *(const v4u*)(qp + 16 + koffA);
            for (int nt = 0; nt < 4; ++nt) {
                FragAB kf;
                const unsigned short* kp =
                    Kbas + (size_t)(key0 + nt * 16 + lh) * DD + dd + koffB;
                kf.u4[0] = *(const v4u*)(kp);
                kf.u4[1] = *(const v4u*)(kp + 16);
                s[nt] = __builtin_amdgcn_wmma_f32_16x16x32_bf16(
                    false, qa.v, false, kf.v, (short)0, s[nt], false, false);
            }
        }
        for (int nt = 0; nt < 4; ++nt)
            for (int i = 0; i < 8; ++i)
                atomicAdd(&Sred[i + 8 * hi][nt * 16 + lh], s[nt][i]);
        __syncthreads();

        // ---- online softmax (all waves redundantly; identical values) --
        float pv[4][8];
        for (int i = 0; i < 8; ++i) {
            const int rowg = q0 + i + 8 * hi;       // query index (local t)
            float mx = -__builtin_inff();
            for (int nt = 0; nt < 4; ++nt) {
                float sv = Sred[i + 8 * hi][nt * 16 + lh];
                const int col = key0 + nt * 16 + lh; // key index
                sv = (col <= rowg) ? sv * scale : -__builtin_inff();
                pv[nt][i] = sv;
                mx = fmaxf(mx, sv);
            }
            for (int msk = 8; msk >= 1; msk >>= 1)       // stay within half
                mx = fmaxf(mx, __shfl_xor(mx, msk, 32));
            const float mnew  = fmaxf(mrow[i], mx);
            const float alpha = (mrow[i] == -__builtin_inff())
                                    ? 0.0f : __expf(mrow[i] - mnew);
            float rs = 0.0f;
            for (int nt = 0; nt < 4; ++nt) {
                float e = __expf(pv[nt][i] - mnew);
                pv[nt][i] = e;
                rs += e;
            }
            for (int msk = 8; msk >= 1; msk >>= 1)
                rs += __shfl_xor(rs, msk, 32);
            lrow[i] = lrow[i] * alpha + rs;
            mrow[i] = mnew;
            for (int nt2 = 0; nt2 < 8; ++nt2) o[nt2][i] *= alpha;
        }
        if (w == 0) {   // one wave publishes P as bf16
            for (int nt = 0; nt < 4; ++nt)
                for (int i = 0; i < 8; ++i)
                    Pb[i + 8 * hi][nt * 16 + lh] = f2bf(pv[nt][i]);
        }
        __syncthreads();

        // ---- O += P @ V (V pre-transposed: contiguous B fragments) -----
        for (int ks = 0; ks < 2; ++ks) {
            FragAB pa;
            const unsigned short* pp = &Pb[lh][ks * 32 + koffA];
            pa.u4[0] = *(const v4u*)(pp);
            pa.u4[1] = *(const v4u*)(pp + 16);
            for (int nt2 = 0; nt2 < 8; ++nt2) {
                FragAB vf;
                const unsigned short* vp =
                    Vbas + (size_t)(d0w + nt2 * 16 + lh) * TT +
                    key0 + ks * 32 + koffB;
                vf.u4[0] = *(const v4u*)(vp);
                vf.u4[1] = *(const v4u*)(vp + 16);
                o[nt2] = __builtin_amdgcn_wmma_f32_16x16x32_bf16(
                    false, pa.v, false, vf.v, (short)0, o[nt2], false, false);
            }
        }
        // next iteration's Sred-zero barrier separates Pb reads from rewrites
    }

    // ---- epilogue: O / l, fp32 out ------------------------------------
    for (int nt2 = 0; nt2 < 8; ++nt2)
        for (int i = 0; i < 8; ++i) {
            const int t = q0 + i + 8 * hi;
            const int d = d0w + nt2 * 16 + lh;
            out[((size_t)(b * TT + t)) * DD + d] = o[nt2][i] / lrow[i];
        }
}

// ---------------------------------------------------------------------------
extern "C" void kernel_launch(void* const* d_in, const int* in_sizes, int n_in,
                              void* d_out, int out_size, void* d_ws, size_t ws_size,
                              hipStream_t stream) {
    (void)in_sizes; (void)n_in; (void)out_size; (void)ws_size;
    const float* x  = (const float*)d_in[0];
    const float* Wq = (const float*)d_in[1];
    const float* Wk = (const float*)d_in[2];
    const float* Wv = (const float*)d_in[3];
    float* out = (float*)d_out;

    unsigned short* Qb = (unsigned short*)d_ws;            // 16 MB
    unsigned short* Kb = Qb + (size_t)BT * DD;             // 16 MB
    unsigned short* Vt = Kb + (size_t)BT * DD;             // 16 MB (transposed)

    // QKV projections: M=8192, N=1024 tiled 128x128, z = {Q,K,V}
    qkv_gemm_kernel<<<dim3(BT / 128, DD / 128, 3), 256, 0, stream>>>(
        x, Wq, Wk, Wv, Qb, Kb, Vt);

    // Flash-style causal attention: one WG per 16 query rows
    causal_attn_kernel<<<dim3(BT / 16), 256, 0, stream>>>(Qb, Kb, Vt, out);
}
